// Encoder_13778255086230
// MI455X (gfx1250) — compile-verified, matrix operand order checked by reading, not run on hardware
//
#include <hip/hip_runtime.h>
#include <hip/hip_bf16.h>

#define BB 64
#define TT 256
#define HH 1024
#define EE 1024
#define NL 2
#define FOURH 4096
#define K2 2048          // concat K: [x (1024) | h (1024)]
#define KCHUNK 128       // K staged per LDS buffer
#define NSTAGE (K2 / KCHUNK)   // 16 stages; 0..7 = x, 8..15 = h
#define LDS_STRIDE 136   // 128 bf16 + 8 pad
#define SGS 132          // f32 gate scratch stride (128 + 4 pad)
#define SMEM_BYTES 34816 // max(2*64*136*2, 64*132*4)

typedef __attribute__((ext_vector_type(16))) __bf16 v16bf;
typedef __attribute__((ext_vector_type(8)))  __bf16 v8bf;
typedef __attribute__((ext_vector_type(8)))  float  v8f;

// fast transcendentals: prefer native CDNA5 TRANS ops when the builtin exists
#if __has_builtin(__builtin_amdgcn_tanhf)
#define FAST_TANH(x) __builtin_amdgcn_tanhf(x)
#elif __has_builtin(__builtin_amdgcn_tanh_f32)
#define FAST_TANH(x) __builtin_amdgcn_tanh_f32(x)
#else
#define FAST_TANH(x) tanhf(x)
#endif

#if __has_builtin(__builtin_amdgcn_rcpf)
#define FAST_RCP(x) __builtin_amdgcn_rcpf(x)
#else
#define FAST_RCP(x) (1.0f / (x))
#endif

__device__ __forceinline__ float fast_sigmoid(float x) {
  return FAST_RCP(1.0f + __expf(-x));
}

// ---------------- prep kernels ----------------

__global__ __launch_bounds__(256) void embed_kernel(const int* __restrict__ src,
                                                    const float* __restrict__ emb,
                                                    __bf16* __restrict__ xemb) {
  size_t idx = (size_t)blockIdx.x * 256 + threadIdx.x;   // over B*T*E
  int e = (int)(idx & (EE - 1));
  size_t bt = idx >> 10;
  int tok = src[bt];
  xemb[idx] = (__bf16)emb[(size_t)tok * EE + e];
}

// pack weights bf16 [L, 4096, 2048]; packed row n' = 4*j + gate  (gate-interleaved)
__global__ __launch_bounds__(256) void pack_kernel(const float* __restrict__ Wih,
                                                   const float* __restrict__ Whh,
                                                   __bf16* __restrict__ Wp) {
  size_t idx = (size_t)blockIdx.x * 256 + threadIdx.x;   // over NL*4096*2048
  int k  = (int)(idx & (K2 - 1));
  int np = (int)((idx >> 11) & (FOURH - 1));             // packed row
  int l  = (int)(idx >> 23);
  int j  = np >> 2;
  int g  = np & 3;
  size_t srow = (size_t)(l * FOURH + g * HH + j);
  float v = (k < HH) ? Wih[srow * HH + k] : Whh[srow * HH + (k - HH)];
  Wp[idx] = (__bf16)v;
}

__global__ __launch_bounds__(256) void bias_kernel(const float* __restrict__ bih,
                                                   const float* __restrict__ bhh,
                                                   float* __restrict__ bias) {
  int idx = blockIdx.x * 256 + threadIdx.x;              // NL*4096
  int l  = idx >> 12;
  int np = idx & (FOURH - 1);
  int j  = np >> 2;
  int g  = np & 3;
  int s  = l * FOURH + g * HH + j;
  bias[idx] = bih[s] + bhh[s];
}

__global__ __launch_bounds__(256) void init_kernel(__bf16* __restrict__ hstate0,
                                                   float* __restrict__ cstate) {
  int idx = blockIdx.x * 256 + threadIdx.x;              // NL*B*H
  hstate0[idx] = (__bf16)0.0f;
  cstate[idx] = 0.0f;
}

// -------- fused recurrent step: gates GEMM (WMMA) + LSTM cell --------
// grid.x = 32 ; block = 256 (8 waves). Wave owns ONE packed-N tile, ALL FOUR M-tiles.
// Block owns packed columns [blk*128, +128) = units [blk*32, +32) for all 64 batch rows.

__device__ __forceinline__ void stage_load(v8bf r[4], int s,
                                           const __bf16* __restrict__ xbase, int xRowStride,
                                           const __bf16* __restrict__ hread,
                                           int srow, int scol) {
  const __bf16* base;
  if (s < NSTAGE / 2) {
    base = xbase + (size_t)srow * xRowStride + s * KCHUNK + scol;
  } else {
    base = hread + (size_t)srow * HH + (s - NSTAGE / 2) * KCHUNK + scol;
  }
#pragma unroll
  for (int i = 0; i < 4; ++i) r[i] = ((const v8bf*)base)[i];
}

__device__ __forceinline__ v16bf load_b(const __bf16* p) {
  v8bf lo = *(const v8bf*)p;
  v8bf hi = *(const v8bf*)(p + 16);
  return __builtin_shufflevector(lo, hi, 0,1,2,3,4,5,6,7,8,9,10,11,12,13,14,15);
}

__device__ __forceinline__ v16bf load_a(const __bf16* lds, int arow, int kk, int half) {
  v8bf lo = *(const v8bf*)&lds[arow * LDS_STRIDE + kk + half * 8];
  v8bf hi = *(const v8bf*)&lds[arow * LDS_STRIDE + kk + 16 + half * 8];
  return __builtin_shufflevector(lo, hi, 0,1,2,3,4,5,6,7,8,9,10,11,12,13,14,15);
}

__global__ __launch_bounds__(256) void lstm_step(const __bf16* __restrict__ xbase, int xRowStride,
                                                 const __bf16* __restrict__ hread,
                                                 const __bf16* __restrict__ Wp_l,
                                                 const float* __restrict__ bias_l,
                                                 float* __restrict__ cstate_l,
                                                 __bf16* __restrict__ hwrite,
                                                 __bf16* __restrict__ hout_bf16, // layer0: h0buf + t*H
                                                 float*  __restrict__ hout_f32,  // layer1: enc + t*H
                                                 float* __restrict__ hn_out,
                                                 float* __restrict__ cn_out,
                                                 int write_final) {
  __shared__ __attribute__((aligned(16))) char smem[SMEM_BYTES];
  __bf16* abuf0 = (__bf16*)smem;                 // A chunk buffer 0 (64 x 128 bf16)
  __bf16* abuf1 = abuf0 + 64 * LDS_STRIDE;       // A chunk buffer 1
  float*  sg    = (float*)smem;                  // aliased f32 gate scratch (after K loop)

  const int tid  = threadIdx.x;
  const int wave = tid >> 5;
  const int lane = tid & 31;
  const int half = lane >> 4;
  const int lrow = lane & 15;

  const int ncol = blockIdx.x * 128 + wave * 16 + lrow;    // packed gate column
  const __bf16* wrow = Wp_l + (size_t)ncol * K2 + half * 8;

  const int srow = tid >> 2;                   // staging: 64 rows x 4 segs of 32 bf16
  const int scol = (tid & 3) * 32;

  v8f acc0 = {}, acc1 = {}, acc2 = {}, acc3 = {};

  v8bf rcur[4], rnext[4];
  stage_load(rcur, 0, xbase, xRowStride, hread, srow, scol);
#pragma unroll
  for (int i = 0; i < 4; ++i)
    ((v8bf*)&abuf0[srow * LDS_STRIDE + scol])[i] = rcur[i];
  stage_load(rnext, 1, xbase, xRowStride, hread, srow, scol);

  v16bf bfrag = load_b(wrow);                  // pipelined weight fragment (k = 0)

#pragma unroll
  for (int s = 0; s < NSTAGE; ++s) {           // fully unrolled: x/h select + parity fold away
    __bf16* bufc = (s & 1) ? abuf1 : abuf0;
    __bf16* bufn = (s & 1) ? abuf0 : abuf1;
    __syncthreads();   // bufc ready; all reads of bufn finished -> refill it
    if (s + 1 < NSTAGE) {
#pragma unroll
      for (int i = 0; i < 4; ++i)
        ((v8bf*)&bufn[srow * LDS_STRIDE + scol])[i] = rnext[i];
    }
    if (s + 2 < NSTAGE)
      stage_load(rnext, s + 2, xbase, xRowStride, hread, srow, scol);

    if (s + 1 < NSTAGE)
      __builtin_prefetch((const void*)(wrow + (s + 1) * KCHUNK), 0, 1);

    // A fragments for kk = 0 of this stage (one LDS bubble per stage, not per K-step)
    v16bf a0 = load_a(bufc,  0 + lrow, 0, half);
    v16bf a1 = load_a(bufc, 16 + lrow, 0, half);
    v16bf a2 = load_a(bufc, 32 + lrow, 0, half);
    v16bf a3 = load_a(bufc, 48 + lrow, 0, half);

#pragma unroll
    for (int kk = 0; kk < KCHUNK; kk += 32) {
      const int kg = s * KCHUNK + kk;
      // pipeline next weight fragment under this step's WMMAs
      v16bf bnext = bfrag;
      if (kg + 32 < K2) bnext = load_b(wrow + kg + 32);

      // pipeline next A fragments (distinct regs; current set stays live)
      v16bf n0 = a0, n1 = a1, n2 = a2, n3 = a3;
      if (kk + 32 < KCHUNK) {
        n0 = load_a(bufc,  0 + lrow, kk + 32, half);
        n1 = load_a(bufc, 16 + lrow, kk + 32, half);
        n2 = load_a(bufc, 32 + lrow, kk + 32, half);
        n3 = load_a(bufc, 48 + lrow, kk + 32, half);
      }

      acc0 = __builtin_amdgcn_wmma_f32_16x16x32_bf16(false, a0, false, bfrag, (short)0, acc0, false, false);
      acc1 = __builtin_amdgcn_wmma_f32_16x16x32_bf16(false, a1, false, bfrag, (short)0, acc1, false, false);
      acc2 = __builtin_amdgcn_wmma_f32_16x16x32_bf16(false, a2, false, bfrag, (short)0, acc2, false, false);
      acc3 = __builtin_amdgcn_wmma_f32_16x16x32_bf16(false, a3, false, bfrag, (short)0, acc3, false, false);
      a0 = n0; a1 = n1; a2 = n2; a3 = n3;
      bfrag = bnext;
    }
  }

  // ---- epilogue: D tiles (+bias) -> LDS f32 scratch ----
  __syncthreads();                 // done with A buffers; alias as f32 gate scratch
  const float bv = bias_l[ncol];
  const int lcol = wave * 16 + lrow;           // 0..127 local packed column
#pragma unroll
  for (int mt = 0; mt < 4; ++mt) {
    const v8f& acc = (mt == 0) ? acc0 : (mt == 1) ? acc1 : (mt == 2) ? acc2 : acc3;
    const int row0 = mt * 16 + half * 8;       // D layout: vgpr r -> M = row0 + r
#pragma unroll
    for (int r = 0; r < 8; ++r)
      sg[(row0 + r) * SGS + lcol] = acc[r] + bv;
  }
  __syncthreads();

  // ---- fused LSTM cell: 64 batch x 32 units per block ----
#pragma unroll
  for (int rr = 0; rr < 8; ++rr) {
    int elem = rr * 256 + tid;                 // 0..2047
    int b = elem >> 5;
    int u = elem & 31;
    int J = blockIdx.x * 32 + u;               // global hidden unit
    float4 g4 = *(const float4*)&sg[b * SGS + 4 * u];   // i, f, g, o (packed order)
    float si = fast_sigmoid(g4.x);
    float sf = fast_sigmoid(g4.y);
    float so = fast_sigmoid(g4.w);
    size_t cidx = (size_t)b * HH + J;
    float c = sf * cstate_l[cidx] + si * FAST_TANH(g4.z);
    float h = so * FAST_TANH(c);
    cstate_l[cidx] = c;
    hwrite[cidx] = (__bf16)h;
    size_t outIdx = (size_t)b * (TT * HH) + J; // bases pre-offset by t*H
    if (hout_bf16) hout_bf16[outIdx] = (__bf16)h;
    if (hout_f32)  hout_f32[outIdx]  = h;
    if (write_final) { hn_out[cidx] = h; cn_out[cidx] = c; }
  }
}

// ---------------- host ----------------
extern "C" void kernel_launch(void* const* d_in, const int* in_sizes, int n_in,
                              void* d_out, int out_size, void* d_ws, size_t ws_size,
                              hipStream_t stream) {
  const int*   src = (const int*)d_in[0];
  const float* emb = (const float*)d_in[1];
  const float* Wih = (const float*)d_in[2];
  const float* Whh = (const float*)d_in[3];
  const float* bih = (const float*)d_in[4];
  const float* bhh = (const float*)d_in[5];
  float* out = (float*)d_out;

  char* ws = (char*)d_ws;
  size_t off = 0;
  auto take = [&](size_t bytes) { size_t p = off; off += (bytes + 255) & ~(size_t)255; return p; };
  __bf16* xemb   = (__bf16*)(ws + take((size_t)BB * TT * EE * 2));
  __bf16* h0buf  = (__bf16*)(ws + take((size_t)BB * TT * HH * 2));
  __bf16* Wp     = (__bf16*)(ws + take((size_t)NL * FOURH * K2 * 2));
  float*  bias   = (float*)(ws + take((size_t)NL * FOURH * 4));
  __bf16* hstate = (__bf16*)(ws + take((size_t)2 * NL * BB * HH * 2)); // ping-pong [2][L][B][H]
  float*  cstate = (float*)(ws + take((size_t)NL * BB * HH * 4));

  embed_kernel<<<(BB * TT * EE) / 256, 256, 0, stream>>>(src, emb, xemb);
  pack_kernel<<<(NL * FOURH * K2) / 256, 256, 0, stream>>>(Wih, Whh, Wp);
  bias_kernel<<<(NL * FOURH) / 256, 256, 0, stream>>>(bih, bhh, bias);
  init_kernel<<<(NL * BB * HH) / 256, 256, 0, stream>>>(hstate, cstate); // zero parity-0 h, all c

  float* enc = out;                          // [B,T,H]
  float* hn  = out + (size_t)BB * TT * HH;   // [L,B,H]
  float* cn  = hn + (size_t)NL * BB * HH;    // [L,B,H]
  const size_t LBH = (size_t)NL * BB * HH;

  for (int t = 0; t < TT; ++t) {
    int rp = t & 1;                          // read parity
    for (int l = 0; l < NL; ++l) {
      const __bf16* xbase = (l == 0) ? (xemb + (size_t)t * EE) : (h0buf + (size_t)t * HH);
      int xRowStride = (l == 0) ? (TT * EE) : (TT * HH);
      const __bf16* hread = hstate + (size_t)rp * LBH + (size_t)l * BB * HH;
      __bf16* hwrite = hstate + (size_t)(rp ^ 1) * LBH + (size_t)l * BB * HH;
      lstm_step<<<32, 256, 0, stream>>>(xbase, xRowStride, hread,
                                        Wp + (size_t)l * FOURH * K2,
                                        bias + (size_t)l * FOURH,
                                        cstate + (size_t)l * BB * HH,
                                        hwrite,
                                        (l == 0) ? (h0buf + (size_t)t * HH) : nullptr,
                                        (l == 1) ? (enc + (size_t)t * HH) : nullptr,
                                        hn + (size_t)l * BB * HH,
                                        cn + (size_t)l * BB * HH,
                                        (t == TT - 1) ? 1 : 0);
    }
  }
}
